// NuModel_17317308137514
// MI455X (gfx1250) — matrix operand order repaired and checked by reading, not compiled, not run on hardware
//
#include <hip/hip_runtime.h>
#include <cstdint>
#include <cstddef>

typedef __attribute__((ext_vector_type(16))) __bf16 v16bf;
typedef __attribute__((ext_vector_type(8)))  __bf16 v8bf;
typedef __attribute__((ext_vector_type(8)))  float  v8f;

#define EPSV 1e-5f

// ---------------------------------------------------------------------------
// Pack edge inputs: A0e[E,32] = [x[row](9) | x[col](9) | edge_attr(12) | 0,0]
// Also writes the x[row] part (+zero padding) of A1e[E,96], accumulates column
// sum/sumsq of the 30 concat features (for BN0 of edge MLP), and in-degree
// counts per destination node.
// ---------------------------------------------------------------------------
__global__ __launch_bounds__(256) void pack_edges_kernel(
    const float* __restrict__ x, const float* __restrict__ ea,
    const int* __restrict__ ei, int E,
    __bf16* __restrict__ A0e, __bf16* __restrict__ A1e,
    float* __restrict__ se_sum, float* __restrict__ se_ssq,
    float* __restrict__ cntN)
{
    __shared__ float ls[32], lq[32];
    int t = threadIdx.x;
    if (t < 32) { ls[t] = 0.f; lq[t] = 0.f; }
    __syncthreads();
    int e = blockIdx.x * 256 + t;
    if (e < E) {
        int r = ei[e], c = ei[E + e];
        float v[30];
        #pragma unroll
        for (int k = 0; k < 9; ++k) v[k] = x[r * 9 + k];
        #pragma unroll
        for (int k = 0; k < 9; ++k) v[9 + k] = x[c * 9 + k];
        #pragma unroll
        for (int k = 0; k < 12; ++k) v[18 + k] = ea[(size_t)e * 12 + k];
        size_t b0 = (size_t)e * 32;
        #pragma unroll
        for (int k = 0; k < 30; ++k) A0e[b0 + k] = (__bf16)v[k];
        A0e[b0 + 30] = (__bf16)0.f; A0e[b0 + 31] = (__bf16)0.f;
        size_t b1 = (size_t)e * 96;
        #pragma unroll
        for (int k = 0; k < 9; ++k) A1e[b1 + k] = (__bf16)v[k];
        #pragma unroll
        for (int k = 9; k < 16; ++k) A1e[b1 + k] = (__bf16)0.f;
        #pragma unroll
        for (int k = 80; k < 96; ++k) A1e[b1 + k] = (__bf16)0.f;
        #pragma unroll
        for (int k = 0; k < 30; ++k) { atomicAdd(&ls[k], v[k]); atomicAdd(&lq[k], v[k] * v[k]); }
        atomicAdd(&cntN[c], 1.0f);
    }
    __syncthreads();
    if (t < 30) { atomicAdd(&se_sum[t], ls[t]); atomicAdd(&se_ssq[t], lq[t]); }
}

// Copy the 9 x-feature stats (over E gathered rows) into the node1-stage stats
// slot (packed positions 0..8); positions 16..79 are filled by the GEMM that
// produces e.
__global__ void copy_stats9_kernel(const float* __restrict__ ss, const float* __restrict__ sq,
                                   float* __restrict__ ds, float* __restrict__ dq)
{
    int t = threadIdx.x;
    if (t < 9) { ds[t] = ss[t]; dq[t] = sq[t]; }
}

// ---------------------------------------------------------------------------
// Pack node inputs: A2n[N,96] = [x(9) | pad | agg(64) | pad], agg = segment
// mean of m over destination nodes. Accumulates stats for node2 BN0 and graph
// member counts.
// ---------------------------------------------------------------------------
__global__ __launch_bounds__(256) void pack_nodes_kernel(
    const float* __restrict__ x, const float* __restrict__ aggsum,
    const float* __restrict__ cntN, const int* __restrict__ batch, int N,
    __bf16* __restrict__ A2n,
    float* __restrict__ sn_sum, float* __restrict__ sn_ssq,
    float* __restrict__ gcnt)
{
    __shared__ float ls[96], lq[96];
    int t = threadIdx.x;
    for (int k = t; k < 96; k += 256) { ls[k] = 0.f; lq[k] = 0.f; }
    __syncthreads();
    int n = blockIdx.x * 256 + t;
    if (n < N) {
        float cnt = cntN[n]; cnt = cnt > 1.f ? cnt : 1.f;
        float inv = 1.f / cnt;
        size_t b = (size_t)n * 96;
        #pragma unroll
        for (int k = 0; k < 9; ++k) {
            float xv = x[n * 9 + k];
            A2n[b + k] = (__bf16)xv;
            atomicAdd(&ls[k], xv); atomicAdd(&lq[k], xv * xv);
        }
        #pragma unroll
        for (int k = 9; k < 16; ++k) A2n[b + k] = (__bf16)0.f;
        for (int k = 0; k < 64; ++k) {
            float av = aggsum[(size_t)n * 64 + k] * inv;
            A2n[b + 16 + k] = (__bf16)av;
            atomicAdd(&ls[16 + k], av); atomicAdd(&lq[16 + k], av * av);
        }
        #pragma unroll
        for (int k = 80; k < 96; ++k) A2n[b + k] = (__bf16)0.f;
        atomicAdd(&gcnt[batch[n]], 1.0f);
    }
    __syncthreads();
    for (int k = t; k < 96; k += 256) { atomicAdd(&sn_sum[k], ls[k]); atomicAdd(&sn_ssq[k], lq[k]); }
}

// Finalize per-graph mean, emit bf16 input to glob MLP, accumulate its stats.
__global__ __launch_bounds__(256) void uprep_kernel(
    const float* __restrict__ gsum, const float* __restrict__ gcnt,
    __bf16* __restrict__ A3, float* __restrict__ sg_sum, float* __restrict__ sg_ssq)
{
    int i = blockIdx.x * 256 + threadIdx.x;   // i < G*64
    int g = i >> 6, c = i & 63;
    float cnt = gcnt[g]; cnt = cnt > 1.f ? cnt : 1.f;
    float v = gsum[i] / cnt;
    A3[i] = (__bf16)v;
    atomicAdd(&sg_sum[c], v);
    atomicAdd(&sg_ssq[c], v * v);
}

// ---------------------------------------------------------------------------
// Fold BatchNorm into weights and emit B in WMMA bf16 fragment layout:
// Bpack[(nn*KK+kk)*32 + lane][16 halves]; element i of lane's v16bf is
// K = kk*32 + 16*(lane>=16) + i, N = nn*16 + (lane&15).
// Packed-K -> raw weight row map: pk<kA -> pk ; kBo<=pk<kBo+kBl -> pk-kBo+rawB
// ---------------------------------------------------------------------------
__device__ __forceinline__ int kmap(int pk, int kA, int kBo, int kBl, int rawB)
{
    if (pk < kA) return pk;
    if (pk >= kBo && pk < kBo + kBl) return pk - kBo + rawB;
    return -1;
}

__global__ void fold_w_kernel(
    const float* __restrict__ w_raw, const float* __restrict__ bn_g, const float* __restrict__ bn_b,
    const float* __restrict__ ssum, const float* __restrict__ sssq, float inv_n,
    int kA, int kBo, int kBl, int rawB, int KK,
    __bf16* __restrict__ Bpack)
{
    int tid = blockIdx.x * blockDim.x + threadIdx.x;
    int total = 4 * KK * 32 * 8;
    if (tid >= total) return;
    int v    = tid & 7;
    int lane = (tid >> 3) & 31;
    int frag = tid >> 8;            // nn*KK + kk
    int kk = frag % KK, nn = frag / KK;
    int col   = nn * 16 + (lane & 15);
    int kbase = kk * 32 + ((lane >> 4) ? 16 : 0);
    #pragma unroll
    for (int h = 0; h < 2; ++h) {
        int i  = 2 * v + h;
        int pk = kbase + i;
        int raw = kmap(pk, kA, kBo, kBl, rawB);
        float val = 0.f;
        if (raw >= 0) {
            float m   = ssum[pk] * inv_n;
            float var = sssq[pk] * inv_n - m * m;
            float s   = bn_g[raw] * rsqrtf(var + EPSV);
            val = s * w_raw[(size_t)raw * 64 + col];
        }
        Bpack[((size_t)frag * 32 + lane) * 16 + i] = (__bf16)val;
    }
}

__global__ void fold_b_kernel(
    const float* __restrict__ w_raw, const float* __restrict__ bn_g, const float* __restrict__ bn_b,
    const float* __restrict__ b_lin,
    const float* __restrict__ ssum, const float* __restrict__ sssq, float inv_n,
    int kA, int kBo, int kBl, int rawB, int KK,
    float* __restrict__ bfold)
{
    int n = threadIdx.x;
    if (n >= 64) return;
    float acc = b_lin[n];
    int Kp = KK * 32;
    for (int pk = 0; pk < Kp; ++pk) {
        int raw = kmap(pk, kA, kBo, kBl, rawB);
        if (raw < 0) continue;
        float m   = ssum[pk] * inv_n;
        float var = sssq[pk] * inv_n - m * m;
        float s   = bn_g[raw] * rsqrtf(var + EPSV);
        acc += (bn_b[raw] - m * s) * w_raw[(size_t)raw * 64 + n];
    }
    bfold[n] = acc;
}

// ---------------------------------------------------------------------------
// Core GEMM: per wave one 16-row x 64-col tile, K = KK*32 via WMMA bf16.
// Fully templated so the epilogue is straight-line (no per-element branching).
// OUTMODE: 0 bf16 store (chaining), 1 f32 atomic scatter (segment-sum),
//          2 f32 store.
// ---------------------------------------------------------------------------
template <int KK, int OUTMODE, bool RELU, bool STATS>
__global__ __launch_bounds__(256) void gemm_wmma(
    const __bf16* __restrict__ A, int lda, int Mtiles,
    const __bf16* __restrict__ Bpack, const float* __restrict__ bias,
    __bf16* __restrict__ out_b, int out_stride, int out_coloff,
    float* __restrict__ out_sc, const int* __restrict__ sidx_arr,
    float* __restrict__ out_f,
    float* __restrict__ stats_sum, float* __restrict__ stats_ssq)
{
    __shared__ float ls[64], lq[64];
    const int lane = threadIdx.x & 31;
    const int wave = threadIdx.x >> 5;
    if (STATS && threadIdx.x < 64) { ls[threadIdx.x] = 0.f; lq[threadIdx.x] = 0.f; }
    if (STATS) __syncthreads();

    int tile = blockIdx.x * 8 + wave;
    bool active = tile < Mtiles;
    int row0 = tile * 16;
    int arow = row0 + (lane & 15);
    int kb   = (lane >> 4) * 16;    // A-fragment K sub-offset per half-wave

    v16bf afrag[KK];
    if (active) {
        // prefetch the A rows this wave will touch 128 tiles later
        __builtin_prefetch(A + ((size_t)arow + 128 * 16) * lda, 0, 1);
        #pragma unroll
        for (int kk = 0; kk < KK; ++kk) {
            const __bf16* p = A + (size_t)arow * lda + kk * 32 + kb;
            v8bf lo = *(const v8bf*)p;          // K = kk*32 + kb + 0..7
            v8bf hi = *(const v8bf*)(p + 8);    // K = kk*32 + kb + 8..15
            afrag[kk] = __builtin_shufflevector(lo, hi,
                           0, 1, 2, 3, 4, 5, 6, 7, 8, 9, 10, 11, 12, 13, 14, 15);
        }
    }
    int sidx[8];
    if (OUTMODE == 1 && active) {
        #pragma unroll
        for (int r = 0; r < 8; ++r) sidx[r] = sidx_arr[row0 + r + (lane >> 4) * 8];
    }

    #pragma unroll
    for (int nn = 0; nn < 4; ++nn) {
        if (active) {
            v8f c;
            #pragma unroll
            for (int r = 0; r < 8; ++r) c[r] = 0.f;
            #pragma unroll
            for (int kk = 0; kk < KK; ++kk) {
                v16bf b = *(const v16bf*)(Bpack + ((size_t)(nn * KK + kk) * 32 + lane) * 16);
                c = __builtin_amdgcn_wmma_f32_16x16x32_bf16(false, afrag[kk], false, b,
                                                            (short)0, c, false, false);
            }
            int col = nn * 16 + (lane & 15);
            float bc = bias[col];
            float s = 0.f, s2 = 0.f;
            int rbase = row0 + (lane >> 4) * 8;
            __bf16* ob = (OUTMODE == 0)
                       ? out_b + (size_t)rbase * out_stride + out_coloff + col : nullptr;
            float*  of = (OUTMODE == 2)
                       ? out_f + (size_t)rbase * 64 + col : nullptr;
            #pragma unroll
            for (int r = 0; r < 8; ++r) {
                float v = c[r] + bc;
                if (RELU) v = v > 0.f ? v : 0.f;
                s += v; s2 += v * v;
                if (OUTMODE == 0) { *ob = (__bf16)v; ob += out_stride; }
                else if (OUTMODE == 1) atomicAdd(&out_sc[(size_t)sidx[r] * 64 + col], v);
                else { *of = v; of += 64; }
            }
            if (STATS) { atomicAdd(&ls[col], s); atomicAdd(&lq[col], s2); }
        }
    }
    if (STATS) {
        __syncthreads();
        if (threadIdx.x < 64) {
            atomicAdd(&stats_sum[threadIdx.x], ls[threadIdx.x]);
            atomicAdd(&stats_ssq[threadIdx.x], lq[threadIdx.x]);
        }
    }
}

// ---------------------------------------------------------------------------
// Heads: 7 branches (sign sigmoid, 6x softmax-4) from u[G,64] -> out[G,25]
// ---------------------------------------------------------------------------
struct HeadP { const float* w[7]; const float* b[7]; };

__global__ __launch_bounds__(64) void heads_kernel(
    const float* __restrict__ u, HeadP hp, float* __restrict__ out)
{
    __shared__ float us[64];
    __shared__ float lg[25];
    int g = blockIdx.x, t = threadIdx.x;
    us[t] = u[(size_t)g * 64 + t];
    __syncthreads();
    if (t < 25) {
        int hi, j, od;
        if (t == 0) { hi = 0; j = 0; od = 1; }
        else        { hi = 1 + (t - 1) / 4; j = (t - 1) & 3; od = 4; }
        float acc = hp.b[hi][j];
        const float* w = hp.w[hi];
        for (int k = 0; k < 64; ++k) acc += us[k] * w[k * od + j];
        lg[t] = acc;
    }
    __syncthreads();
    if (t < 25) {
        float o;
        if (t == 0) {
            o = 1.f / (1.f + expf(-lg[0]));
        } else {
            int b0 = 1 + ((t - 1) / 4) * 4;
            float m = fmaxf(fmaxf(lg[b0], lg[b0 + 1]), fmaxf(lg[b0 + 2], lg[b0 + 3]));
            float den = expf(lg[b0] - m) + expf(lg[b0 + 1] - m) +
                        expf(lg[b0 + 2] - m) + expf(lg[b0 + 3] - m);
            o = expf(lg[t] - m) / den;
        }
        out[(size_t)g * 25 + t] = o;
    }
}

// ---------------------------------------------------------------------------
// Host orchestration
// ---------------------------------------------------------------------------
extern "C" void kernel_launch(void* const* d_in, const int* in_sizes, int n_in,
                              void* d_out, int out_size, void* d_ws, size_t ws_size,
                              hipStream_t stream)
{
    const float* x     = (const float*)d_in[0];
    const float* ea    = (const float*)d_in[1];
    const int*   ei    = (const int*)d_in[2];
    const int*   batch = (const int*)d_in[3];
    const int N = in_sizes[0] / 9;
    const int E = in_sizes[1] / 12;
    const int G = out_size / 25;

    struct Mlp { const float *bg[3], *bb[3], *w[3], *b[3]; };
    auto getM = [&](int base) {
        Mlp m;
        for (int i = 0; i < 3; ++i) {
            m.bg[i] = (const float*)d_in[base + 4 * i + 0];
            m.bb[i] = (const float*)d_in[base + 4 * i + 1];
            m.w [i] = (const float*)d_in[base + 4 * i + 2];
            m.b [i] = (const float*)d_in[base + 4 * i + 3];
        }
        return m;
    };
    Mlp Pe = getM(4), Pn1 = getM(16), Pn2 = getM(28), Pg = getM(40);
    HeadP hp;
    for (int h = 0; h < 7; ++h) {
        hp.w[h] = (const float*)d_in[52 + 2 * h];
        hp.b[h] = (const float*)d_in[53 + 2 * h];
    }

    // ---- workspace carve-out -------------------------------------------
    char* base = (char*)d_ws;
    size_t off = 0;
    auto alloc = [&](size_t bytes) -> char* {
        char* p = base + off;
        off = (off + bytes + 255) & ~(size_t)255;
        return p;
    };
    float* stats  = (float*)alloc(16 * 192 * sizeof(float)); // 12 slots used: sum[96]+ssq[96]
    float* aggsum = (float*)alloc((size_t)N * 64 * 4);
    float* cntN   = (float*)alloc((size_t)N * 4);
    float* gsum   = (float*)alloc((size_t)G * 64 * 4);
    float* gcnt   = (float*)alloc((size_t)G * 4);
    size_t zero_bytes = off;                                  // everything above needs zeroing
    __bf16* A0e = (__bf16*)alloc((size_t)E * 32 * 2);
    __bf16* A1e = (__bf16*)alloc((size_t)E * 96 * 2);
    __bf16* hA  = (__bf16*)alloc((size_t)E * 64 * 2);
    __bf16* hB  = (__bf16*)alloc((size_t)E * 64 * 2);
    __bf16* A2n = (__bf16*)alloc((size_t)N * 96 * 2);
    __bf16* A3  = (__bf16*)alloc((size_t)G * 64 * 2);
    __bf16* hGa = (__bf16*)alloc((size_t)G * 64 * 2);
    __bf16* hGb = (__bf16*)alloc((size_t)G * 64 * 2);
    float*  uf  = (float*)alloc((size_t)G * 64 * 4);
    __bf16* Bp[12];
    float*  bf_[12];
    for (int i = 0; i < 12; ++i) Bp[i]  = (__bf16*)alloc(4 * 3 * 32 * 16 * 2);
    for (int i = 0; i < 12; ++i) bf_[i] = (float*)alloc(64 * 4);

    hipMemsetAsync(d_ws, 0, zero_bytes, stream);

    auto S = [&](int slot) { return stats + slot * 192; };   // +0 sum, +96 sumsq
    const float invE = 1.f / (float)E, invN = 1.f / (float)N, invG = 1.f / (float)G;

    auto fold = [&](const Mlp& m, int li, float* st, float inv_n,
                    int kA, int kBo, int kBl, int rawB, int KK, int slot) {
        int total = 4 * KK * 256;
        fold_w_kernel<<<(total + 255) / 256, 256, 0, stream>>>(
            m.w[li], m.bg[li], m.bb[li], st, st + 96, inv_n, kA, kBo, kBl, rawB, KK, Bp[slot]);
        fold_b_kernel<<<1, 64, 0, stream>>>(
            m.w[li], m.bg[li], m.bb[li], m.b[li], st, st + 96, inv_n, kA, kBo, kBl, rawB, KK, bf_[slot]);
    };
    auto grid_for = [](int Mtiles) { return dim3((Mtiles + 7) / 8); };
    const dim3 blk(256);

    // ---- Edge stage -----------------------------------------------------
    pack_edges_kernel<<<(E + 255) / 256, 256, 0, stream>>>(x, ea, ei, E, A0e, A1e,
                                                           S(0), S(0) + 96, cntN);
    copy_stats9_kernel<<<1, 32, 0, stream>>>(S(0), S(0) + 96, S(3), S(3) + 96);

    fold(Pe, 0, S(0), invE, 30, 0, 0, 0, 1, 0);
    gemm_wmma<1, 0, true, true><<<grid_for(E / 16), blk, 0, stream>>>(
        A0e, 32, E / 16, Bp[0], bf_[0], hA, 64, 0, nullptr, nullptr, nullptr, S(1), S(1) + 96);
    fold(Pe, 1, S(1), invE, 64, 0, 0, 0, 2, 1);
    gemm_wmma<2, 0, true, true><<<grid_for(E / 16), blk, 0, stream>>>(
        hA, 64, E / 16, Bp[1], bf_[1], hB, 64, 0, nullptr, nullptr, nullptr, S(2), S(2) + 96);
    fold(Pe, 2, S(2), invE, 64, 0, 0, 0, 2, 2);
    // e -> A1e cols 16..79 ; its stats land at packed positions 16..79 of slot 3
    gemm_wmma<2, 0, false, true><<<grid_for(E / 16), blk, 0, stream>>>(
        hB, 64, E / 16, Bp[2], bf_[2], A1e, 96, 16, nullptr, nullptr, nullptr,
        S(3) + 16, S(3) + 96 + 16);

    // ---- Node MLP1 over edges + scatter-mean ---------------------------
    fold(Pn1, 0, S(3), invE, 9, 16, 64, 9, 3, 3);
    gemm_wmma<3, 0, true, true><<<grid_for(E / 16), blk, 0, stream>>>(
        A1e, 96, E / 16, Bp[3], bf_[3], hA, 64, 0, nullptr, nullptr, nullptr, S(4), S(4) + 96);
    fold(Pn1, 1, S(4), invE, 64, 0, 0, 0, 2, 4);
    gemm_wmma<2, 0, true, true><<<grid_for(E / 16), blk, 0, stream>>>(
        hA, 64, E / 16, Bp[4], bf_[4], hB, 64, 0, nullptr, nullptr, nullptr, S(5), S(5) + 96);
    fold(Pn1, 2, S(5), invE, 64, 0, 0, 0, 2, 5);
    gemm_wmma<2, 1, false, false><<<grid_for(E / 16), blk, 0, stream>>>(
        hB, 64, E / 16, Bp[5], bf_[5], nullptr, 0, 0, aggsum, ei + E, nullptr, nullptr, nullptr);

    // ---- Node MLP2 over nodes + scatter-mean to graphs -----------------
    pack_nodes_kernel<<<(N + 255) / 256, 256, 0, stream>>>(x, aggsum, cntN, batch, N, A2n,
                                                           S(6), S(6) + 96, gcnt);
    fold(Pn2, 0, S(6), invN, 9, 16, 64, 9, 3, 6);
    gemm_wmma<3, 0, true, true><<<grid_for(N / 16), blk, 0, stream>>>(
        A2n, 96, N / 16, Bp[6], bf_[6], hA, 64, 0, nullptr, nullptr, nullptr, S(7), S(7) + 96);
    fold(Pn2, 1, S(7), invN, 64, 0, 0, 0, 2, 7);
    gemm_wmma<2, 0, true, true><<<grid_for(N / 16), blk, 0, stream>>>(
        hA, 64, N / 16, Bp[7], bf_[7], hB, 64, 0, nullptr, nullptr, nullptr, S(8), S(8) + 96);
    fold(Pn2, 2, S(8), invN, 64, 0, 0, 0, 2, 8);
    gemm_wmma<2, 1, false, false><<<grid_for(N / 16), blk, 0, stream>>>(
        hB, 64, N / 16, Bp[8], bf_[8], nullptr, 0, 0, gsum, batch, nullptr, nullptr, nullptr);

    // ---- Global MLP -----------------------------------------------------
    uprep_kernel<<<(G * 64) / 256, 256, 0, stream>>>(gsum, gcnt, A3, S(9), S(9) + 96);
    fold(Pg, 0, S(9), invG, 64, 0, 0, 0, 2, 9);
    gemm_wmma<2, 0, true, true><<<grid_for(G / 16), blk, 0, stream>>>(
        A3, 64, G / 16, Bp[9], bf_[9], hGa, 64, 0, nullptr, nullptr, nullptr, S(10), S(10) + 96);
    fold(Pg, 1, S(10), invG, 64, 0, 0, 0, 2, 10);
    gemm_wmma<2, 0, true, true><<<grid_for(G / 16), blk, 0, stream>>>(
        hGa, 64, G / 16, Bp[10], bf_[10], hGb, 64, 0, nullptr, nullptr, nullptr, S(11), S(11) + 96);
    fold(Pg, 2, S(11), invG, 64, 0, 0, 0, 2, 11);
    gemm_wmma<2, 2, false, false><<<grid_for(G / 16), blk, 0, stream>>>(
        hGb, 64, G / 16, Bp[11], bf_[11], nullptr, 0, 0, nullptr, nullptr, uf, nullptr, nullptr);

    // ---- Heads ----------------------------------------------------------
    heads_kernel<<<G, 64, 0, stream>>>(uf, hp, (float*)d_out);
}
